// SelfAttention_33337536151577
// MI455X (gfx1250) — matrix operand order repaired and checked by reading, not compiled, not run on hardware
//
#include <hip/hip_runtime.h>

typedef __attribute__((ext_vector_type(16))) __bf16 v16bf;
typedef __attribute__((ext_vector_type(8)))  float  v8f;
typedef __attribute__((ext_vector_type(4)))  int    v4i;

typedef __attribute__((address_space(1))) v4i* gv4i_p;   // global int4*
typedef __attribute__((address_space(3))) v4i* lv4i_p;   // LDS int4*

#define B_    4
#define S_    4096
#define D_    64
#define KT    64           // KV tile rows (4 x 16-key WMMA blocks)
#define WAVES 4
#define QTB   (WAVES * 16) // 64 query rows per workgroup

#if __has_builtin(__builtin_amdgcn_global_load_async_to_lds_b128)
#define USE_ASYNC 1
#else
#define USE_ASYNC 0
#endif

// ---- helpers -------------------------------------------------------------

__device__ __forceinline__ unsigned short f2bf(float f) {
  __bf16 h = (__bf16)f;                 // native v_cvt (RNE) on gfx1250
  return __builtin_bit_cast(unsigned short, h);
}

// raw v_exp_f32: args are always <= 0 here, so libm's |x|>126 fixup sequence
// (cmp/cndmask/ldexp) is dead weight — underflow-to-zero is the wanted tail.
__device__ __forceinline__ float fast_exp2(float x) {
#if __has_builtin(__builtin_amdgcn_exp2f)
  return __builtin_amdgcn_exp2f(x);
#else
  return exp2f(x);
#endif
}

// max/sum across the 16 lanes of each half-wave (xor distances 1,2,4,8 stay
// inside a 16-lane group). ds_swizzle group-of-32 imm: {xor[14:10],or[9:5],and[4:0]}
__device__ __forceinline__ float hmax16(float x) {
  x = fmaxf(x, __int_as_float(__builtin_amdgcn_ds_swizzle(__float_as_int(x), 0x041F)));
  x = fmaxf(x, __int_as_float(__builtin_amdgcn_ds_swizzle(__float_as_int(x), 0x081F)));
  x = fmaxf(x, __int_as_float(__builtin_amdgcn_ds_swizzle(__float_as_int(x), 0x101F)));
  x = fmaxf(x, __int_as_float(__builtin_amdgcn_ds_swizzle(__float_as_int(x), 0x201F)));
  return x;
}
__device__ __forceinline__ float hsum16(float x) {
  x += __int_as_float(__builtin_amdgcn_ds_swizzle(__float_as_int(x), 0x041F));
  x += __int_as_float(__builtin_amdgcn_ds_swizzle(__float_as_int(x), 0x081F));
  x += __int_as_float(__builtin_amdgcn_ds_swizzle(__float_as_int(x), 0x101F));
  x += __int_as_float(__builtin_amdgcn_ds_swizzle(__float_as_int(x), 0x201F));
  return x;
}

union Frag {
  v16bf bf;
  uint4 q[2];
};

// one 16-byte global -> LDS transfer (async, ASYNCcnt-tracked, on gfx1250)
__device__ __forceinline__ void copy16(const unsigned short* g, unsigned short* l) {
#if USE_ASYNC
  __builtin_amdgcn_global_load_async_to_lds_b128((gv4i_p)(g), (lv4i_p)(l), 0, 0);
#else
  *(uint4*)l = *(const uint4*)g;
#endif
}

__device__ __forceinline__ void wait_async() {
#if USE_ASYNC
#if __has_builtin(__builtin_amdgcn_s_wait_asynccnt)
  __builtin_amdgcn_s_wait_asynccnt(0);
#else
  asm volatile("s_wait_asynccnt 0x0" ::: "memory");
#endif
#endif
}

// ---- pre-pass: fp32 -> bf16; Q pre-scaled by (1/sqrt(D))*log2(e) so the
// softmax can run natively in base-2 (v_exp_f32 is exp2). V transposed. ----

__global__ void cvt_kernel(const float* __restrict__ Q, const float* __restrict__ K,
                           const float* __restrict__ V,
                           unsigned short* __restrict__ Qb,
                           unsigned short* __restrict__ Kb,
                           unsigned short* __restrict__ VT) {
  int idx = blockIdx.x * blockDim.x + threadIdx.x;
  const int total = B_ * S_ * D_;
  if (idx >= total) return;
  int d = idx % D_;
  int s = (idx / D_) % S_;
  int b = idx / (D_ * S_);
  Qb[idx] = f2bf(Q[idx] * 0.18033688f);   // 0.125 * log2(e)
  Kb[idx] = f2bf(K[idx]);
  VT[(size_t)b * D_ * S_ + (size_t)d * S_ + s] = f2bf(V[idx]);
}

// ---- main flash-attention kernel ----------------------------------------

__global__ __launch_bounds__(WAVES * 32)
void fa_kernel(const unsigned short* __restrict__ Qb,
               const unsigned short* __restrict__ Kb,
               const unsigned short* __restrict__ VT,
               float* __restrict__ O) {
  // double-buffered K / V^T tiles + per-wave P transpose buffer (40 KB total)
  __shared__ __align__(16) unsigned short lK[2][KT][D_];   // 2 x 8 KB
  __shared__ __align__(16) unsigned short lV[2][D_][KT];   // 2 x 8 KB (V^T: [d][k])
  __shared__ __align__(16) unsigned short lP[WAVES][16][KT]; // 8 KB

  const int tid  = threadIdx.x;
  const int w    = tid >> 5;
  const int lane = tid & 31;
  const int half = lane >> 4;     // 0 = lanes 0-15, 1 = lanes 16-31
  const int l15  = lane & 15;

  const int nQB       = S_ / QTB;
  const int b         = blockIdx.x / nQB;
  const int qBlock    = (blockIdx.x % nQB) * QTB;
  const int qWaveBase = qBlock + w * 16;

  const unsigned short* gK = Kb + (size_t)b * S_ * D_;
  const unsigned short* gV = VT + (size_t)b * D_ * S_;

  // ---- load Q fragments (A-matrix layout, 16x32 bf16, two k=32 chunks) ----
  Frag aQ[2];
  {
    const unsigned short* qrow = Qb + ((size_t)b * S_ + (size_t)(qWaveBase + l15)) * D_;
#pragma unroll
    for (int c = 0; c < 2; ++c) {
      const int chB = c * 32 + half * 8;
      aQ[c].q[0] = *(const uint4*)(qrow + chB);
      aQ[c].q[1] = *(const uint4*)(qrow + chB + 16);
    }
  }

  v8f   Oacc[4];
  float rowMax[8], rowSum[8];
#pragma unroll
  for (int n = 0; n < 4; ++n) Oacc[n] = (v8f){0.f, 0.f, 0.f, 0.f, 0.f, 0.f, 0.f, 0.f};
#pragma unroll
  for (int r = 0; r < 8; ++r) { rowMax[r] = -__builtin_inff(); rowSum[r] = 0.f; }

  // ---- cooperative tile staging (global -> LDS async, double-buffered) ----
  // K tile: 64x64 bf16 contiguous = 512 uint4; V^T tile: 64 rows x 8 uint4.
  auto stage = [&](int buf, int kvBase) {
    unsigned short* dK = &lK[buf][0][0];
    unsigned short* dV = &lV[buf][0][0];
    const unsigned short* srcK = gK + (size_t)kvBase * D_;
#pragma unroll
    for (int it = 0; it < 4; ++it) {
      const int j = tid + it * 128;
      copy16(srcK + j * 8, dK + j * 8);
    }
#pragma unroll
    for (int it = 0; it < 4; ++it) {
      const int j    = tid + it * 128;
      const int dch  = j >> 3;
      const int part = j & 7;
      copy16(gV + (size_t)dch * S_ + kvBase + part * 8, dV + j * 8);
    }
  };

  const int nTiles = qBlock / KT + 1;   // covers keys 0 .. qBlock+63

  stage(0, 0);
  wait_async();
  __syncthreads();

  for (int t = 0; t < nTiles; ++t) {
    const int kvBase = t * KT;
    const int buf    = t & 1;

    // prefetch next tile into the other buffer while we compute this one
    if (t + 1 < nTiles) stage(buf ^ 1, (t + 1) * KT);

    // ---- S' = (Q*scale*log2e) K^T : four 16-key blocks, K=64 in two k=32 chunks
    v8f Sacc[4];
#pragma unroll
    for (int j = 0; j < 4; ++j) {
      const unsigned short* krow = &lK[buf][j * 16 + l15][0];
      const int chB = half * 8;
      Frag bK0, bK1;
      bK0.q[0] = *(const uint4*)(krow + chB);
      bK0.q[1] = *(const uint4*)(krow + chB + 16);
      bK1.q[0] = *(const uint4*)(krow + 32 + chB);
      bK1.q[1] = *(const uint4*)(krow + 32 + chB + 16);
      v8f acc = (v8f){0.f, 0.f, 0.f, 0.f, 0.f, 0.f, 0.f, 0.f};
      acc = __builtin_amdgcn_wmma_f32_16x16x32_bf16(false, aQ[0].bf, false, bK0.bf,
                                                    (short)0, acc, false, false);
      acc = __builtin_amdgcn_wmma_f32_16x16x32_bf16(false, aQ[1].bf, false, bK1.bf,
                                                    (short)0, acc, false, false);
      Sacc[j] = acc;
    }

    // ---- causal mask: only ever needed on the final tile (kvBase == qBlock) ----
    if (t == nTiles - 1) {
#pragma unroll
      for (int r = 0; r < 8; ++r) {
        const int qRow = qWaveBase + r + half * 8;
#pragma unroll
        for (int j = 0; j < 4; ++j) {
          const int key = kvBase + j * 16 + l15;
          Sacc[j][r] = (key > qRow) ? -1e30f : Sacc[j][r];
        }
      }
    }

    // ---- online softmax in base-2 (v_exp_f32 is native exp2) ----
    float alpha[8];
#pragma unroll
    for (int r = 0; r < 8; ++r) {
      const float m = hmax16(fmaxf(fmaxf(Sacc[0][r], Sacc[1][r]),
                                   fmaxf(Sacc[2][r], Sacc[3][r])));
      const float mNew = fmaxf(rowMax[r], m);
      alpha[r]  = fast_exp2(rowMax[r] - mNew);  // first iter: exp2(-inf)=0
      rowMax[r] = mNew;
    }
    float tSum[8];
#pragma unroll
    for (int r = 0; r < 8; ++r) {
      const int m = r + half * 8;
      float s = 0.f;
#pragma unroll
      for (int j = 0; j < 4; ++j) {
        const float p = fast_exp2(Sacc[j][r] - rowMax[r]);
        s += p;
        lP[w][m][j * 16 + l15] = f2bf(p);
      }
      tSum[r] = s;
    }
#pragma unroll
    for (int r = 0; r < 8; ++r)
      rowSum[r] = rowSum[r] * alpha[r] + hsum16(tSum[r]);
#pragma unroll
    for (int n = 0; n < 4; ++n)
#pragma unroll
      for (int r = 0; r < 8; ++r)
        Oacc[n][r] = Oacc[n][r] * alpha[r];

    // LDS is in-order per wave; make the cross-lane P exchange visible
    asm volatile("s_wait_dscnt 0x0" ::: "memory");

    // ---- O += P V : A = P (16x64, two k=32 fragments), B = V^T columns ----
#pragma unroll
    for (int kc = 0; kc < 2; ++kc) {
      Frag aP;
      {
        const unsigned short* prow = &lP[w][l15][0] + kc * 32;
        const int kB = half * 8;
        aP.q[0] = *(const uint4*)(prow + kB);
        aP.q[1] = *(const uint4*)(prow + kB + 16);
      }
#pragma unroll
      for (int n = 0; n < 4; ++n) {
        const unsigned short* vrow = &lV[buf][n * 16 + l15][0] + kc * 32;
        Frag bV;
        bV.q[0] = *(const uint4*)(vrow + half * 8);
        bV.q[1] = *(const uint4*)(vrow + half * 8 + 16);
        Oacc[n] = __builtin_amdgcn_wmma_f32_16x16x32_bf16(false, aP.bf, false, bV.bf,
                                                          (short)0, Oacc[n], false, false);
      }
    }

    wait_async();        // my staging transfers for tile t+1 have landed in LDS
    __syncthreads();     // everyone done reading buf / done staging buf^1
  }

  // ---- epilogue: O / rowSum, fp32 out ----
  float* orow = O + (size_t)b * S_ * D_;
#pragma unroll
  for (int r = 0; r < 8; ++r) {
    const int   m   = r + half * 8;
    const float inv = 1.0f / rowSum[r];
    const size_t base = (size_t)(qWaveBase + m) * D_;
#pragma unroll
    for (int n = 0; n < 4; ++n)
      orow[base + n * 16 + l15] = Oacc[n][r] * inv;
  }
}

// ---- host entry ----------------------------------------------------------

extern "C" void kernel_launch(void* const* d_in, const int* in_sizes, int n_in,
                              void* d_out, int out_size, void* d_ws, size_t ws_size,
                              hipStream_t stream) {
  const float* Q = (const float*)d_in[0];
  const float* K = (const float*)d_in[1];
  const float* V = (const float*)d_in[2];
  float* Out = (float*)d_out;

  const size_t nElem = (size_t)B_ * S_ * D_;
  unsigned short* Qb = (unsigned short*)d_ws;
  unsigned short* Kb = Qb + nElem;
  unsigned short* VT = Kb + nElem;

  const int cvtBlocks = (int)((nElem + 255) / 256);
  cvt_kernel<<<cvtBlocks, 256, 0, stream>>>(Q, K, V, Qb, Kb, VT);

  const int grid = B_ * (S_ / QTB);
  fa_kernel<<<grid, WAVES * 32, 0, stream>>>(Qb, Kb, VT, Out);
}
